// Correlation_2757369004065
// MI455X (gfx1250) — matrix operand order repaired and checked by reading, not compiled, not run on hardware
//
#include <hip/hip_runtime.h>

typedef __attribute__((ext_vector_type(16))) _Float16 v16h;
typedef __attribute__((ext_vector_type(8)))  _Float16 v8h;
typedef __attribute__((ext_vector_type(8)))  float    v8f;

#define B_   8
#define C_   128
#define H_   48
#define W_   64
#define HW   3072                 // H*W
#define OUT_PER_B (HW * HW)       // 9437184
#define LDS_STRIDE 136            // 128 halves + 8 pad: b128 frag reads hit all 64 banks

// ---------------- zero column-sum buffer ----------------
__global__ void zero_f32(float* __restrict__ p, int n) {
  int i = blockIdx.x * blockDim.x + threadIdx.x;
  if (i < n) p[i] = 0.0f;
}

// ---------------- transpose + f32->f16 convert ----------------
// src: f32 (B, C, H, W) natural layout (spatial flat n' = u*W + v).
// dst: f16 [b][m][c], c contiguous (row stride 128 halves).
// is_x != 0: m = v*H + u  (channel permutation of the reference); else m = n'.
__global__ void transpose_cvt(const float* __restrict__ src,
                              _Float16* __restrict__ dst, int is_x) {
  __shared__ unsigned tile[32 * 65];   // [s][c-pair], stride 65 dwords (conflict-free)
  const int t   = threadIdx.x;
  const int b   = blockIdx.x / 96;
  const int sp0 = (blockIdx.x % 96) * 32;

  {  // phase 1: coalesced reads along spatial; pack 2 channels into one dword
    const int s   = t & 31;
    const int cp0 = t >> 5;                    // 0..7
    const float* sb = src + (size_t)b * C_ * HW + (sp0 + s);
#pragma unroll
    for (int p = 0; p < 8; ++p) {
      const int cp = p * 8 + cp0;              // 0..63  (c = 2*cp)
      const int c  = cp * 2;
      float f0 = sb[(size_t)c * HW];
      float f1 = sb[(size_t)(c + 1) * HW];
      union { _Float16 h[2]; unsigned u; } pk;
      pk.h[0] = (_Float16)f0;
      pk.h[1] = (_Float16)f1;
      tile[s * 65 + cp] = pk.u;
    }
  }
  __syncthreads();
  {  // phase 2: coalesced writes along c
    const int c2 = t & 63;                     // dword index (c = 2*c2)
    const int s0 = t >> 6;                     // 0..3
    unsigned* du = (unsigned*)dst;
#pragma unroll
    for (int p = 0; p < 8; ++p) {
      const int s  = p * 4 + s0;               // 0..31
      const int np = sp0 + s;
      int m;
      if (is_x) { int u = np >> 6; int v = np & 63; m = v * H_ + u; }
      else      { m = np; }
      du[(size_t)(b * HW + m) * 64 + c2] = tile[s * 65 + c2];
    }
  }
}

// ---------------- shared WMMA tile core ----------------
// Tile: 128(M) x 64(N), full K=128 staged in LDS. 8 waves: 4(M) x 2(N),
// each wave 32x32 via 2x2 f32 accumulators of v_wmma_f32_16x16x32_f16.
// Caller owns the epilogue; acc layout per ISA: lane l, vgpr r ->
// row = 16*mf + r + 8*(l>=16), col = 16*nf + (l&15).
__device__ __forceinline__ void corr_tile(const _Float16* __restrict__ xt,
                                          const _Float16* __restrict__ yt,
                                          int b, int m0, int n0,
                                          v8f acc[2][2]) {
  __shared__ __align__(16) _Float16 As[128 * LDS_STRIDE];
  __shared__ __align__(16) _Float16 Bs[64  * LDS_STRIDE];
  const int t = threadIdx.x;

  {  // stage tiles (contiguous b128 copies from the pre-transposed f16 buffers)
    const v8h* xg = (const v8h*)(xt + (size_t)(b * HW + m0) * 128);
#pragma unroll
    for (int i = 0; i < 8; ++i) {
      int f = i * 256 + t;                 // 0..2047 (8-half chunks)
      v8h v = xg[f];
      *(v8h*)&As[(f >> 4) * LDS_STRIDE + (f & 15) * 8] = v;
    }
    const v8h* yg = (const v8h*)(yt + (size_t)(b * HW + n0) * 128);
#pragma unroll
    for (int i = 0; i < 4; ++i) {
      int f = i * 256 + t;                 // 0..1023
      v8h v = yg[f];
      *(v8h*)&Bs[(f >> 4) * LDS_STRIDE + (f & 15) * 8] = v;
    }
  }
  __syncthreads();

  const int lane = t & 31;
  const int wid  = t >> 5;
  const int wm   = wid & 3;                // M sub-tile: 32 rows
  const int wn   = wid >> 2;               // N sub-tile: 32 cols
  const int l16  = lane & 15;
  const int hi   = lane >> 4;
  const int sA   = hi * 8;                 // A frag K-offset per lane half
  const int sB   = hi * 8;                 // B frag K-offset per lane half

#pragma unroll
  for (int kb = 0; kb < 4; ++kb) {         // K = 4 * 32
    v16h av[2], bv[2];
#pragma unroll
    for (int mf = 0; mf < 2; ++mf) {
      const _Float16* p = &As[(wm * 32 + mf * 16 + l16) * LDS_STRIDE + kb * 32];
      v8h a0 = *(const v8h*)(p + sA);
      v8h a1 = *(const v8h*)(p + 16 + sA);
      av[mf] = __builtin_shufflevector(a0, a1, 0,1,2,3,4,5,6,7,
                                               8,9,10,11,12,13,14,15);
    }
#pragma unroll
    for (int nf = 0; nf < 2; ++nf) {
      const _Float16* p = &Bs[(wn * 32 + nf * 16 + l16) * LDS_STRIDE + kb * 32];
      v8h b0 = *(const v8h*)(p + sB);
      v8h b1 = *(const v8h*)(p + 16 + sB);
      bv[nf] = __builtin_shufflevector(b0, b1, 0,1,2,3,4,5,6,7,
                                               8,9,10,11,12,13,14,15);
    }
#pragma unroll
    for (int mf = 0; mf < 2; ++mf)
#pragma unroll
      for (int nf = 0; nf < 2; ++nf)
        acc[mf][nf] = __builtin_amdgcn_wmma_f32_16x16x32_f16(
            false, av[mf], false, bv[nf], (short)0, acc[mf][nf], false, false);
  }
}

// ---------------- pass 1: column sums of relu(corr)^2 (no output stores) ----
__global__ void __launch_bounds__(256)
sumsq_gemm(const _Float16* __restrict__ xt, const _Float16* __restrict__ yt,
           float* __restrict__ colsum) {
  __shared__ float red[64];
  const int t  = threadIdx.x;
  const int b  = blockIdx.z;
  const int m0 = blockIdx.y * 128;
  const int n0 = blockIdx.x * 64;

  if (t < 64) red[t] = 0.0f;     // covered by the staging barrier in corr_tile

  v8f acc[2][2] = {};
  corr_tile(xt, yt, b, m0, n0, acc);

  const int lane = t & 31;
  const int wid  = t >> 5;
  const int wn   = wid >> 2;
  const int l16  = lane & 15;

#pragma unroll
  for (int nf = 0; nf < 2; ++nf) {
    float sq = 0.0f;
#pragma unroll
    for (int mf = 0; mf < 2; ++mf)
#pragma unroll
      for (int r = 0; r < 8; ++r) {
        float v = fmaxf(acc[mf][nf][r], 0.0f);
        sq += v * v;
      }
    atomicAdd(&red[wn * 32 + nf * 16 + l16], sq);   // ds_add_f32
  }
  __syncthreads();
  if (t < 64) atomicAdd(&colsum[b * HW + n0 + t], red[t]);  // 64 global atomics/WG
}

// ---------------- pass 2: recompute, normalize, store ----------------
__global__ void __launch_bounds__(256)
norm_gemm(const _Float16* __restrict__ xt, const _Float16* __restrict__ yt,
          const float* __restrict__ colsum, float* __restrict__ out) {
  const int t  = threadIdx.x;
  const int b  = blockIdx.z;
  const int m0 = blockIdx.y * 128;
  const int n0 = blockIdx.x * 64;

  v8f acc[2][2] = {};
  corr_tile(xt, yt, b, m0, n0, acc);

  const int lane = t & 31;
  const int wid  = t >> 5;
  const int wm   = wid & 3;
  const int wn   = wid >> 2;
  const int l16  = lane & 15;
  const int hi   = lane >> 4;

  float* outb = out + (size_t)b * OUT_PER_B;
#pragma unroll
  for (int nf = 0; nf < 2; ++nf) {
    const int n = n0 + wn * 32 + nf * 16 + l16;
    const float inv = rsqrtf(colsum[b * HW + n] + 1e-6f);
#pragma unroll
    for (int mf = 0; mf < 2; ++mf) {
      const int mbase = m0 + wm * 32 + mf * 16 + hi * 8;
#pragma unroll
      for (int r = 0; r < 8; ++r) {
        float v = fmaxf(acc[mf][nf][r], 0.0f) * inv;
        outb[(size_t)(mbase + r) * HW + n] = v;
      }
    }
  }
}

extern "C" void kernel_launch(void* const* d_in, const int* in_sizes, int n_in,
                              void* d_out, int out_size, void* d_ws, size_t ws_size,
                              hipStream_t stream) {
  const float* x = (const float*)d_in[0];
  const float* y = (const float*)d_in[1];
  float* out = (float*)d_out;

  // workspace layout (12.68 MB total):
  _Float16* xt = (_Float16*)d_ws;                              // 6,291,456 B
  _Float16* yt = (_Float16*)((char*)d_ws + 6291456);           // 6,291,456 B
  float* colsum = (float*)((char*)d_ws + 12582912);            //    98,304 B

  zero_f32<<<96, 256, 0, stream>>>(colsum, B_ * HW);
  transpose_cvt<<<B_ * 96, 256, 0, stream>>>(x, xt, 1);
  transpose_cvt<<<B_ * 96, 256, 0, stream>>>(y, yt, 0);
  dim3 grid(HW / 64, HW / 128, B_);
  sumsq_gemm<<<grid, 256, 0, stream>>>(xt, yt, colsum);
  norm_gemm <<<grid, 256, 0, stream>>>(xt, yt, colsum, out);
}
